// FeatureDGCNN_128849019260
// MI455X (gfx1250) — matrix operand order repaired and checked by reading, not compiled, not run on hardware
//
#include <hip/hip_runtime.h>
#include <hip/hip_bf16.h>
#include <cstdint>

typedef __attribute__((ext_vector_type(16))) _Float16 v16h;
typedef __attribute__((ext_vector_type(8)))  float    v8f;

#define NPTS  2048
#define BATCH 8
#define KNB   20
#define EPSBN 1e-5f
#define SLOPE 0.2f

__device__ __forceinline__ float atomicMaxF(float* addr, float value) {
    // sign-split trick: int-max for >=0, uint-min for <0 (works with -inf init)
    if (value >= 0.f)
        return __int_as_float(atomicMax((int*)addr, __float_as_int(value)));
    return __uint_as_float(atomicMin((unsigned int*)addr, __float_as_uint(value)));
}

// ---------------------------------------------------------------------------
// kNN: one wave per point. 64 candidate distances per lane held in VGPRs,
// 20 rounds of (local argmax -> shfl_xor cross-lane argmax -> invalidate).
// Tie-break = smaller index, matching jax.lax.top_k.
// ---------------------------------------------------------------------------
__global__ __launch_bounds__(256) void knn_kernel(const float* __restrict__ x,
                                                  int* __restrict__ idx) {
    const int lane = threadIdx.x & 31;
    const int wave = threadIdx.x >> 5;
    const int p = blockIdx.x * 8 + wave;
    const int b = p / NPTS, n = p % NPTS;
    const float* xb = x + (size_t)b * 3 * NPTS;
    const float a0 = xb[n], a1 = xb[NPTS + n], a2 = xb[2 * NPTS + n];
    const float aa = a0 * a0 + a1 * a1 + a2 * a2;

    float d[64];
#pragma unroll
    for (int i = 0; i < 64; ++i) {
        const int m = i * 32 + lane;
        const float p0 = xb[m], p1 = xb[NPTS + m], p2 = xb[2 * NPTS + m];
        const float dot = a0 * p0 + a1 * p1 + a2 * p2;
        d[i] = 2.f * dot - aa - (p0 * p0 + p1 * p1 + p2 * p2);
    }

    int* out = idx + ((size_t)b * NPTS + n) * KNB;
    for (int t = 0; t < KNB; ++t) {
        float best = -__builtin_inff();
        int slot = 0;
#pragma unroll
        for (int i = 0; i < 64; ++i)
            if (d[i] > best) { best = d[i]; slot = i; }
        float bv = best;
        int bi = slot * 32 + lane;
#pragma unroll
        for (int off = 16; off >= 1; off >>= 1) {
            const float ov = __shfl_xor(bv, off, 32);
            const int   oi = __shfl_xor(bi, off, 32);
            if (ov > bv || (ov == bv && oi < bi)) { bv = ov; bi = oi; }
        }
        if (lane == 0) out[t] = bi;
        if ((bi & 31) == lane) {
            const int ws = bi >> 5;
#pragma unroll
            for (int i = 0; i < 64; ++i)
                if (i == ws) d[i] = -__builtin_inff();
        }
    }
}

// ---------------------------------------------------------------------------
// Repack weights (O x Kdim, f32 row-major) into per-(tile,chunk) WMMA A-lane
// layout, f16: Wsw[((tile*chunks + kc)*32 + lane)*16 + e], e = 2*v + j.
// ---------------------------------------------------------------------------
__global__ void repack_weights(const float* __restrict__ W,
                               _Float16* __restrict__ Wsw, int O, int Kdim) {
    const int chunks = (Kdim + 31) >> 5;
    const size_t total = (size_t)(O >> 4) * chunks * 32 * 16;
    size_t t = (size_t)blockIdx.x * blockDim.x + threadIdx.x;
    if (t >= total) return;
    const int e = (int)(t & 15);
    const int l = (int)((t >> 4) & 31);
    size_t r = t >> 9;
    const int kc = (int)(r % chunks);
    const int tile = (int)(r / chunks);
    const int v = e >> 1, j = e & 1;
    const int hf = l >> 4;
    const int row = tile * 16 + (l & 15);
    const int ka = 2 * v + j + 16 * hf;
    const int kg = kc * 32 + ka;
    Wsw[t] = (_Float16)((kg < Kdim) ? W[(size_t)row * Kdim + kg] : 0.f);
}

// ---------------------------------------------------------------------------
// Fill a strided (B, O, N) channel-slice with a constant (for -inf max init).
// ---------------------------------------------------------------------------
__global__ void fill_strided(float* base, size_t bStride, int O, float v) {
    size_t t = (size_t)blockIdx.x * blockDim.x + threadIdx.x;
    const size_t total = (size_t)BATCH * O * NPTS;
    if (t >= total) return;
    const int n = (int)(t % NPTS);
    const size_t r = t / NPTS;
    const int o = (int)(r % O);
    const int b = (int)(r / O);
    base[(size_t)b * bStride + (size_t)o * NPTS + n] = v;
}

// ---------------------------------------------------------------------------
// EdgeConv epilogue helper: BN stats + segmented max-over-k scatter.
// Columns are consecutive (b,n,kk); same-n runs are contiguous within each
// 16-lane group, so a predicated shfl_down suffix-max collapses the run and
// only segment heads (kk==0 or group start) issue the global float atomic.
// ---------------------------------------------------------------------------
__device__ __forceinline__ void ec_epilogue(float val, int o, int OUT,
                                            int lane, int kk,
                                            float* __restrict__ sums,
                                            float* __restrict__ zaddr) {
    float s = val, q = val * val;
#pragma unroll
    for (int off = 1; off < 16; off <<= 1) {
        s += __shfl_xor(s, off, 32);
        q += __shfl_xor(q, off, 32);
    }
    if ((lane & 15) == 0) {
        atomicAdd(&sums[o], s);
        atomicAdd(&sums[OUT + o], q);
    }
    // segmented suffix-max over same-n runs inside the 16-lane group
    float mval = val;
#pragma unroll
    for (int st = 1; st < 16; st <<= 1) {
        const float ov = __shfl_down(mval, st, 32);
        if (((lane & 15) + st < 16) && (kk + st < KNB)) mval = fmaxf(mval, ov);
    }
    if (kk == 0 || (lane & 15) == 0) atomicMaxF(zaddr, mval);
}

// ---------------------------------------------------------------------------
// EdgeConv layer 1 (Cin=3): scalar f32 gather path (K=6, one chunk).
// ---------------------------------------------------------------------------
template <int OUT, int MT>
__global__ __launch_bounds__(256) void edgeconv_wmma_xyz(
        const float* __restrict__ in,
        const _Float16* __restrict__ Wsw,
        const int* __restrict__ knn,
        float* __restrict__ z, size_t zBStride,
        float* __restrict__ sums) {
    constexpr int CIN = 3, KD = 6;
    constexpr int MGROUPS = (OUT / 16) / MT;
    const int lane = threadIdx.x & 31;
    const int wave = threadIdx.x >> 5;
    constexpr int COLS = BATCH * NPTS * KNB;
    constexpr int TILESN = COLS >> 4;
    const int tile = blockIdx.x * 8 + wave;
    if (tile >= MGROUPS * TILESN) return;
    const int tG = tile % MGROUPS, tN = tile / MGROUPS;

    const int col = tN * 16 + (lane & 15);
    const int b  = col / (NPTS * KNB);
    const int r  = col % (NPTS * KNB);
    const int n  = r / KNB;
    const int kk = r % KNB;
    const int m  = knn[((size_t)b * NPTS + n) * KNB + kk];
    const float* inB = in + (size_t)b * 3 * NPTS;
    const int hf = lane >> 4;

    v16h bmat = {};
#pragma unroll
    for (int e = 0; e < 16; ++e) {
        const int cg = hf * 16 + e;
        float fv = 0.f;
        if (cg < CIN)
            fv = inB[(size_t)cg * NPTS + m] - inB[(size_t)cg * NPTS + n];
        else if (cg < KD)
            fv = inB[(size_t)(cg - CIN) * NPTS + n];
        bmat[e] = (_Float16)fv;
    }

    v8f acc[MT];
#pragma unroll
    for (int mt = 0; mt < MT; ++mt) {
        const int tM = tG * MT + mt;
        const v16h amat = *(const v16h*)(Wsw + ((size_t)tM * 32 + lane) * 16);
        acc[mt] = __builtin_amdgcn_wmma_f32_16x16x32_f16(
            false, amat, false, bmat, (short)0, (v8f){}, false, false);
    }

#pragma unroll
    for (int mt = 0; mt < MT; ++mt)
#pragma unroll
        for (int v = 0; v < 8; ++v) {
            const int o = (tG * MT + mt) * 16 + (hf ? (8 + v) : v);
            ec_epilogue(acc[mt][v], o, OUT, lane, kk, sums,
                        &z[(size_t)b * zBStride + (size_t)o * NPTS + n]);
        }
}

// ---------------------------------------------------------------------------
// EdgeConv layers 2-4: input is f16 point-major (B, N, RS) (+channel offset
// pre-applied). B fragment = vector loads of neighbor/center rows + packed
// f16 subtract. Cin must be a multiple of 32. MT M-tiles reuse the fragment.
// ---------------------------------------------------------------------------
template <int CIN, int OUT, int MT, int RS>
__global__ __launch_bounds__(256) void edgeconv_wmma_f16(
        const _Float16* __restrict__ inT,
        const _Float16* __restrict__ Wsw,
        const int* __restrict__ knn,
        float* __restrict__ z, size_t zBStride,
        float* __restrict__ sums) {
    constexpr int KD = 2 * CIN;
    constexpr int CHUNKS = KD / 32;
    constexpr int MGROUPS = (OUT / 16) / MT;
    const int lane = threadIdx.x & 31;
    const int wave = threadIdx.x >> 5;
    constexpr int COLS = BATCH * NPTS * KNB;
    constexpr int TILESN = COLS >> 4;
    const int tile = blockIdx.x * 8 + wave;
    if (tile >= MGROUPS * TILESN) return;
    const int tG = tile % MGROUPS, tN = tile / MGROUPS;

    const int col = tN * 16 + (lane & 15);
    const int b  = col / (NPTS * KNB);
    const int r  = col % (NPTS * KNB);
    const int n  = r / KNB;
    const int kk = r % KNB;
    const int m  = knn[((size_t)b * NPTS + n) * KNB + kk];
    const int hf = lane >> 4;
    const _Float16* rowN = inT + ((size_t)b * NPTS + n) * RS;
    const _Float16* rowM = inT + ((size_t)b * NPTS + m) * RS;

    v8f acc[MT];
#pragma unroll
    for (int mt = 0; mt < MT; ++mt) acc[mt] = (v8f){};

#pragma unroll
    for (int kc = 0; kc < CHUNKS; ++kc) {
        const int base = kc * 32 + hf * 16;   // uniform side: CIN % 32 == 0
        v16h bmat;
        if (base < CIN) {
            const v16h nb = *(const v16h*)(rowM + base);
            const v16h ct = *(const v16h*)(rowN + base);
            bmat = nb - ct;
        } else {
            bmat = *(const v16h*)(rowN + (base - CIN));
        }
#pragma unroll
        for (int mt = 0; mt < MT; ++mt) {
            const int tM = tG * MT + mt;
            const v16h amat = *(const v16h*)(Wsw +
                (((size_t)tM * CHUNKS + kc) * 32 + lane) * 16);
            acc[mt] = __builtin_amdgcn_wmma_f32_16x16x32_f16(
                false, amat, false, bmat, (short)0, acc[mt], false, false);
        }
    }

#pragma unroll
    for (int mt = 0; mt < MT; ++mt)
#pragma unroll
        for (int v = 0; v < 8; ++v) {
            const int o = (tG * MT + mt) * 16 + (hf ? (8 + v) : v);
            ec_epilogue(acc[mt][v], o, OUT, lane, kk, sums,
                        &z[(size_t)b * zBStride + (size_t)o * NPTS + n]);
        }
}

// ---------------------------------------------------------------------------
// Dense 1x1-conv GEMM: (O x C) @ (C x B*N), f16 point-major input (B,N,RS).
// One v16h load per K-chunk per lane; MT M-tiles reuse it. Optional bias /
// BN stats; stores f32 pre-activation y.
// ---------------------------------------------------------------------------
template <int CIN, int OUT, int MT, int RS>
__global__ __launch_bounds__(256) void dense_wmma(
        const _Float16* __restrict__ inT,
        const _Float16* __restrict__ Wsw,
        const float* __restrict__ bias, int hasBias,
        float* __restrict__ out, size_t outBStride,
        float* __restrict__ sums, int doStats) {
    constexpr int CHUNKS = CIN / 32;
    constexpr int MGROUPS = (OUT / 16) / MT;
    const int lane = threadIdx.x & 31;
    const int wave = threadIdx.x >> 5;
    constexpr int COLS = BATCH * NPTS;
    constexpr int TILESN = COLS >> 4;
    const int tile = blockIdx.x * 8 + wave;
    if (tile >= MGROUPS * TILESN) return;
    const int tG = tile % MGROUPS, tN = tile / MGROUPS;

    const int col = tN * 16 + (lane & 15);
    const int b = col / NPTS, n = col % NPTS;
    const int hf = lane >> 4;
    const _Float16* row = inT + ((size_t)b * NPTS + n) * RS;

    v8f acc[MT];
#pragma unroll
    for (int mt = 0; mt < MT; ++mt) acc[mt] = (v8f){};

    for (int kc = 0; kc < CHUNKS; ++kc) {
        const v16h bmat = *(const v16h*)(row + kc * 32 + hf * 16);
#pragma unroll
        for (int mt = 0; mt < MT; ++mt) {
            const int tM = tG * MT + mt;
            const v16h amat = *(const v16h*)(Wsw +
                (((size_t)tM * CHUNKS + kc) * 32 + lane) * 16);
            acc[mt] = __builtin_amdgcn_wmma_f32_16x16x32_f16(
                false, amat, false, bmat, (short)0, acc[mt], false, false);
        }
    }

#pragma unroll
    for (int mt = 0; mt < MT; ++mt) {
#pragma unroll
        for (int v = 0; v < 8; ++v) {
            const int o = (tG * MT + mt) * 16 + (hf ? (8 + v) : v);
            float val = acc[mt][v] + (hasBias ? bias[o] : 0.f);
            if (doStats) {
                float s = val, q = val * val;
#pragma unroll
                for (int off = 1; off < 16; off <<= 1) {
                    s += __shfl_xor(s, off, 32);
                    q += __shfl_xor(q, off, 32);
                }
                if ((lane & 15) == 0) {
                    atomicAdd(&sums[o], s);
                    atomicAdd(&sums[OUT + o], q);
                }
            }
            out[(size_t)b * outBStride + (size_t)o * NPTS + n] = val;
        }
    }
}

// ---------------------------------------------------------------------------
// BatchNorm (training-mode, from accumulated sum/sumsq) + LeakyReLU.
// Reads f32 pre-activation (B,O,N strided), writes f16 point-major (B,N,TS)
// at channel offset chanOff (feeds the next WMMA stage directly).
// ---------------------------------------------------------------------------
__global__ void bn_lrelu_finalize(const float* __restrict__ y, size_t bStride, int O,
                                  const float* __restrict__ sums,
                                  const float* __restrict__ g,
                                  const float* __restrict__ bb, float invCnt,
                                  _Float16* __restrict__ outT, int outTStride,
                                  int chanOff) {
    size_t t = (size_t)blockIdx.x * blockDim.x + threadIdx.x;
    const size_t total = (size_t)BATCH * O * NPTS;
    if (t >= total) return;
    const int n = (int)(t % NPTS);
    const size_t r = t / NPTS;
    const int o = (int)(r % O);
    const int b = (int)(r / O);
    const float mean = sums[o] * invCnt;
    const float var = sums[O + o] * invCnt - mean * mean;
    const float yv = y[(size_t)b * bStride + (size_t)o * NPTS + n];
    float res = (yv - mean) * rsqrtf(var + EPSBN) * g[o] + bb[o];
    res = (res >= 0.f) ? res : SLOPE * res;
    outT[((size_t)b * NPTS + n) * outTStride + chanOff + o] = (_Float16)res;
}

// ---------------------------------------------------------------------------
static inline void repack(const float* W, _Float16* Wsw, int O, int Kdim,
                          hipStream_t stream) {
    const int chunks = (Kdim + 31) >> 5;
    const size_t total = (size_t)(O >> 4) * chunks * 512;
    repack_weights<<<(unsigned)((total + 255) / 256), 256, 0, stream>>>(W, Wsw, O, Kdim);
}

extern "C" void kernel_launch(void* const* d_in, const int* in_sizes, int n_in,
                              void* d_out, int out_size, void* d_ws, size_t ws_size,
                              hipStream_t stream) {
    const float* x  = (const float*)d_in[0];
    const float* W1 = (const float*)d_in[1],  *g1 = (const float*)d_in[2],  *b1 = (const float*)d_in[3];
    const float* W2 = (const float*)d_in[4],  *g2 = (const float*)d_in[5],  *b2 = (const float*)d_in[6];
    const float* W3 = (const float*)d_in[7],  *g3 = (const float*)d_in[8],  *b3 = (const float*)d_in[9];
    const float* W4 = (const float*)d_in[10], *g4 = (const float*)d_in[11], *b4 = (const float*)d_in[12];
    const float* W5 = (const float*)d_in[13], *g5 = (const float*)d_in[14], *b5 = (const float*)d_in[15];
    const float* W6 = (const float*)d_in[16], *c6 = (const float*)d_in[17];
    const float* g6 = (const float*)d_in[18], *b6 = (const float*)d_in[19];
    const float* W7 = (const float*)d_in[20], *c7 = (const float*)d_in[21];
    const float* g7 = (const float*)d_in[22], *b7 = (const float*)d_in[23];
    const float* W8 = (const float*)d_in[24], *c8 = (const float*)d_in[25];

    const size_t N = NPTS, Bsz = BATCH;
    char* ws = (char*)d_ws;
    size_t off = 0;
    int*      idx   = (int*)(ws + off);      off += Bsz * N * KNB * sizeof(int);
    float*    zcat  = (float*)(ws + off);    off += Bsz * 512  * N * sizeof(float); // pre-BN max acc
    float*    h5    = (float*)(ws + off);    off += Bsz * 1024 * N * sizeof(float); // pre-BN y
    float*    h6    = (float*)(ws + off);    off += Bsz * 512  * N * sizeof(float);
    float*    h7    = (float*)(ws + off);    off += Bsz * 128  * N * sizeof(float);
    float*    stats = (float*)(ws + off);    off += 2 * 1024 * sizeof(float);
    _Float16* wsw   = (_Float16*)(ws + off); off += 1024 * 1024 * sizeof(_Float16);
    _Float16* hcatT = (_Float16*)(ws + off); off += Bsz * N * 512  * sizeof(_Float16);
    _Float16* h5T   = (_Float16*)(ws + off); off += Bsz * N * 1024 * sizeof(_Float16);
    _Float16* h6T   = (_Float16*)(ws + off); off += Bsz * N * 512  * sizeof(_Float16);
    _Float16* h7T   = (_Float16*)(ws + off); off += Bsz * N * 128  * sizeof(_Float16);
    (void)ws_size; (void)n_in; (void)in_sizes; (void)out_size;

    constexpr int ecCols = BATCH * NPTS * KNB;   // 327680
    constexpr int dCols  = BATCH * NPTS;         // 16384
    const float ecInv = 1.f / (float)ecCols;
    const float dInv  = 1.f / (float)dCols;

    // 1) kNN graph
    knn_kernel<<<(Bsz * N) / 8, 256, 0, stream>>>(x, idx);

    // 2) EdgeConv layers; pre-BN max into zcat slices, BN+LReLU into hcatT
    auto ec_prep = [&](float* zslice, int O, const float* W, int Kdim) {
        hipMemsetAsync(stats, 0, 2 * O * sizeof(float), stream);
        const size_t elems = Bsz * (size_t)O * N;
        fill_strided<<<(unsigned)((elems + 255) / 256), 256, 0, stream>>>(
            zslice, 512 * N, O, -__builtin_inff());
        repack(W, wsw, O, Kdim, stream);
    };
    auto ec_fini = [&](const float* zslice, int O, const float* g,
                       const float* bb, int chanOff) {
        const size_t elems = Bsz * (size_t)O * N;
        bn_lrelu_finalize<<<(unsigned)((elems + 255) / 256), 256, 0, stream>>>(
            zslice, 512 * N, O, stats, g, bb, ecInv, hcatT, 512, chanOff);
    };

    { // L1: Cin=3 -> 64 (f32 scalar gather, single K chunk)
        float* zs = zcat;
        ec_prep(zs, 64, W1, 6);
        const int tiles = (64 / 16 / 4) * (ecCols / 16);
        edgeconv_wmma_xyz<64, 4><<<(tiles + 7) / 8, 256, 0, stream>>>(
            x, wsw, idx, zs, 512 * N, stats);
        ec_fini(zs, 64, g1, b1, 0);
    }
    { // L2: 64 -> 64
        float* zs = zcat + 64 * N;
        ec_prep(zs, 64, W2, 128);
        const int tiles = (64 / 16 / 4) * (ecCols / 16);
        edgeconv_wmma_f16<64, 64, 4, 512><<<(tiles + 7) / 8, 256, 0, stream>>>(
            hcatT + 0, wsw, idx, zs, 512 * N, stats);
        ec_fini(zs, 64, g2, b2, 64);
    }
    { // L3: 64 -> 128
        float* zs = zcat + 128 * N;
        ec_prep(zs, 128, W3, 128);
        const int tiles = (128 / 16 / 4) * (ecCols / 16);
        edgeconv_wmma_f16<64, 128, 4, 512><<<(tiles + 7) / 8, 256, 0, stream>>>(
            hcatT + 64, wsw, idx, zs, 512 * N, stats);
        ec_fini(zs, 128, g3, b3, 128);
    }
    { // L4: 128 -> 256
        float* zs = zcat + 256 * N;
        ec_prep(zs, 256, W4, 256);
        const int tiles = (256 / 16 / 4) * (ecCols / 16);
        edgeconv_wmma_f16<128, 256, 4, 512><<<(tiles + 7) / 8, 256, 0, stream>>>(
            hcatT + 128, wsw, idx, zs, 512 * N, stats);
        ec_fini(zs, 256, g4, b4, 256);
    }

    // 3) encoder_last + feature head (all inputs f16 point-major)
    { // conv5: 512 -> 1024, BN+LReLU
        hipMemsetAsync(stats, 0, 2 * 1024 * sizeof(float), stream);
        repack(W5, wsw, 1024, 512, stream);
        const int tiles = (1024 / 16 / 4) * (dCols / 16);
        dense_wmma<512, 1024, 4, 512><<<(tiles + 7) / 8, 256, 0, stream>>>(
            hcatT, wsw, nullptr, 0, h5, 1024 * N, stats, 1);
        const size_t elems = Bsz * 1024 * N;
        bn_lrelu_finalize<<<(unsigned)((elems + 255) / 256), 256, 0, stream>>>(
            h5, 1024 * N, 1024, stats, g5, b5, dInv, h5T, 1024, 0);
    }
    { // conv6: 1024 -> 512, +bias, BN+LReLU
        hipMemsetAsync(stats, 0, 2 * 512 * sizeof(float), stream);
        repack(W6, wsw, 512, 1024, stream);
        const int tiles = (512 / 16 / 4) * (dCols / 16);
        dense_wmma<1024, 512, 4, 1024><<<(tiles + 7) / 8, 256, 0, stream>>>(
            h5T, wsw, c6, 1, h6, 512 * N, stats, 1);
        const size_t elems = Bsz * 512 * N;
        bn_lrelu_finalize<<<(unsigned)((elems + 255) / 256), 256, 0, stream>>>(
            h6, 512 * N, 512, stats, g6, b6, dInv, h6T, 512, 0);
    }
    { // conv7: 512 -> 128, +bias, BN+LReLU
        hipMemsetAsync(stats, 0, 2 * 128 * sizeof(float), stream);
        repack(W7, wsw, 128, 512, stream);
        const int tiles = (128 / 16 / 4) * (dCols / 16);
        dense_wmma<512, 128, 4, 512><<<(tiles + 7) / 8, 256, 0, stream>>>(
            h6T, wsw, c7, 1, h7, 128 * N, stats, 1);
        const size_t elems = Bsz * 128 * N;
        bn_lrelu_finalize<<<(unsigned)((elems + 255) / 256), 256, 0, stream>>>(
            h7, 128 * N, 128, stats, g7, b7, dInv, h7T, 128, 0);
    }
    { // conv8: 128 -> 32, +bias only -> d_out (f32)
        repack(W8, wsw, 32, 128, stream);
        const int tiles = (32 / 16 / 2) * (dCols / 16);
        dense_wmma<128, 32, 2, 128><<<(tiles + 7) / 8, 256, 0, stream>>>(
            h7T, wsw, c8, 1, (float*)d_out, 32 * N, stats, 0);
    }
}
